// Seq2Seq_86663850098867
// MI455X (gfx1250) — matrix operand order repaired and checked by reading, not compiled, not run on hardware
//
#include <hip/hip_runtime.h>

typedef __attribute__((ext_vector_type(16))) __bf16 v16bf;
typedef __attribute__((ext_vector_type(8)))  __bf16 v8bf;
typedef __attribute__((ext_vector_type(8)))  float  v8f;

#define HDIM 512
#define G4   2048   // 4*H
#define BSZ  16
#define TLEN 256
#define VOC  32000

__device__ __forceinline__ float sigm(float x) { return 1.0f / (1.0f + __expf(-x)); }
__device__ __forceinline__ float tanh_fast(float x) {
  x = fminf(fmaxf(x, -15.0f), 15.0f);
  float e = __expf(2.0f * x);
  return (e - 1.0f) / (e + 1.0f);
}

// A-operand (16x32 bf16) load: lane = 16*half + row;
// lane holds row M=row, K = k0+16*half .. +15
__device__ __forceinline__ v16bf load_a(const __bf16* base /*16 x HDIM, row-major*/,
                                        int k0, int lane) {
  const int row = lane & 15, half = lane >> 4;
  const v8bf* p = (const v8bf*)(base + row * HDIM + k0 + 16 * half);
  v8bf lo = p[0];
  v8bf hi = p[1];   // +8 elements
  return __builtin_shufflevector(lo, hi, 0,1,2,3,4,5,6,7,8,9,10,11,12,13,14,15);
}

// B-operand (32x16 bf16) load: lane = 16*half + n; holds col N=n, K = k0+16*half..+15.
// W is row-major (N, K) with stride HDIM -> 32 contiguous bytes per lane.
// Pure 32-bit index so the backend can use saddr + i32 voffset (GVS) addressing.
__device__ __forceinline__ v16bf load_b(const __bf16* __restrict__ W,
                                        int nbase, int k0, int lane) {
  const int nl = lane & 15, half = lane >> 4;
  const int idx = (nbase + nl) * HDIM + k0 + 16 * half;
  return *(const v16bf*)(W + idx);
}

// ---------------- prep kernels ----------------

__global__ void cvt_f32_bf16(const float* __restrict__ src, __bf16* __restrict__ dst, int n) {
  int i = blockIdx.x * blockDim.x + threadIdx.x;
  if (i < n) dst[i] = (__bf16)src[i];
}

// xg[b][n] = sum_k embed_W[words[b,0]][k] * w_ih1[n][k] + b_ih1[n] + b_hh1[n]
// (upper half of x is zero, so only K<512 of w_ih1 contributes)
__global__ void xgates_kernel(const int* __restrict__ words, const float* __restrict__ embed_W,
                              const float* __restrict__ w_ih1, const float* __restrict__ b_ih1,
                              const float* __restrict__ b_hh1, float* __restrict__ xg) {
  int idx = blockIdx.x * blockDim.x + threadIdx.x;
  if (idx >= BSZ * G4) return;
  int b = idx / G4, n = idx % G4;
  int word = words[b * TLEN];
  const float* x0 = embed_W + (long long)word * HDIM;
  const float* wr = w_ih1 + (long long)n * (2 * HDIM);
  float s = b_ih1[n] + b_hh1[n];
  for (int k = 0; k < HDIM; ++k) s = fmaf(x0[k], wr[k], s);
  xg[idx] = s;
}

__global__ void bias2_kernel(const float* __restrict__ b_ih2, const float* __restrict__ b_hh2,
                             float* __restrict__ bias2) {
  int n = blockIdx.x * blockDim.x + threadIdx.x;
  if (n < G4) bias2[n] = b_ih2[n] + b_hh2[n];
}

__global__ void zero_t0_kernel(float* __restrict__ out) {
  int i = blockIdx.x * blockDim.x + threadIdx.x;   // BSZ*VOC
  if (i < BSZ * VOC) {
    int b = i / VOC, v = i % VOC;
    __builtin_nontemporal_store(0.0f, &out[(long long)b * TLEN * VOC + v]);
  }
}

// ---------------- persistent recurrence (single WGP) ----------------
// 32 waves; wave w owns columns [16w, 16w+16) of h/c for both layers.
// c1/c2 live in VGPRs across all 255 steps; h passes through LDS in bf16.
// Weight tiles are re-read from (L2-resident) global every step. A laundered
// 32-bit zero index defeats LICM hoisting (which previously spilled ~6KB/lane
// to scratch) while keeping saddr+voffset global addressing.

__global__ __launch_bounds__(1024) void lstm_seq_kernel(
    const __bf16* __restrict__ w_hh1,   // (2048, 512) bf16 row-major
    const __bf16* __restrict__ w_ih2,
    const __bf16* __restrict__ w_hh2,
    const float*  __restrict__ xg,      // (16, 2048)  precomputed layer-1 input gates
    const float*  __restrict__ bias2,   // (2048)      b_ih2 + b_hh2
    __bf16* __restrict__ h2_seq)        // (255, 16, 512) bf16
{
  __shared__ __align__(16) __bf16 h1_s[BSZ * HDIM];
  __shared__ __align__(16) __bf16 h2_s[BSZ * HDIM];

  for (int i = threadIdx.x; i < BSZ * HDIM; i += blockDim.x) {
    h1_s[i] = (__bf16)0.0f; h2_s[i] = (__bf16)0.0f;
  }
  __syncthreads();

  const int lane = threadIdx.x & 31;
  const int wave = threadIdx.x >> 5;     // 0..31
  const int nl = lane & 15, half = lane >> 4;
  const int n0 = wave * 16;

  v8f c1 = {}; v8f c2 = {};
  float b2g[4];
  #pragma unroll
  for (int g = 0; g < 4; ++g) b2g[g] = bias2[g * HDIM + n0 + nl];

  int lv = 0;   // stays 0, but the compiler must assume it varies per step
  for (int t = 0; t < TLEN - 1; ++t) {
    asm volatile("" : "+s"(lv));

    // -------- layer 1: acc = xg + h1_old @ w_hh1^T
    v8f acc[4];
    #pragma unroll
    for (int g = 0; g < 4; ++g)
      #pragma unroll
      for (int r = 0; r < 8; ++r)
        acc[g][r] = xg[(r + 8 * half) * G4 + g * HDIM + n0 + nl + lv];

    for (int k0 = 0; k0 < HDIM; k0 += 32) {
      v16bf a = load_a(h1_s, k0, lane);
      #pragma unroll
      for (int g = 0; g < 4; ++g) {
        v16bf bm = load_b(w_hh1, g * HDIM + n0 + lv, k0, lane);
        acc[g] = __builtin_amdgcn_wmma_f32_16x16x32_bf16(false, a, false, bm,
                                                         (short)0, acc[g], false, false);
      }
    }
    __syncthreads();   // all reads of old h1_s complete

    #pragma unroll
    for (int r = 0; r < 8; ++r) {
      float ig = sigm(acc[0][r]);
      float fg = sigm(acc[1][r]);
      float gg = tanh_fast(acc[2][r]);
      float og = sigm(acc[3][r]);
      float c = fg * c1[r] + ig * gg;
      c1[r] = c;
      float h = og * tanh_fast(c);
      h1_s[(r + 8 * half) * HDIM + n0 + nl] = (__bf16)h;
    }
    __syncthreads();   // new h1_s visible

    // -------- layer 2: acc = bias2 + h1_new @ w_ih2^T + h2_old @ w_hh2^T
    #pragma unroll
    for (int g = 0; g < 4; ++g)
      #pragma unroll
      for (int r = 0; r < 8; ++r)
        acc[g][r] = b2g[g];

    for (int k0 = 0; k0 < HDIM; k0 += 32) {
      v16bf a1 = load_a(h1_s, k0, lane);
      v16bf a2 = load_a(h2_s, k0, lane);
      #pragma unroll
      for (int g = 0; g < 4; ++g) {
        v16bf bi = load_b(w_ih2, g * HDIM + n0 + lv, k0, lane);
        acc[g] = __builtin_amdgcn_wmma_f32_16x16x32_bf16(false, a1, false, bi,
                                                         (short)0, acc[g], false, false);
        v16bf bh = load_b(w_hh2, g * HDIM + n0 + lv, k0, lane);
        acc[g] = __builtin_amdgcn_wmma_f32_16x16x32_bf16(false, a2, false, bh,
                                                         (short)0, acc[g], false, false);
      }
    }
    __syncthreads();   // all reads of old h2_s complete

    #pragma unroll
    for (int r = 0; r < 8; ++r) {
      float ig = sigm(acc[0][r]);
      float fg = sigm(acc[1][r]);
      float gg = tanh_fast(acc[2][r]);
      float og = sigm(acc[3][r]);
      float c = fg * c2[r] + ig * gg;
      c2[r] = c;
      float h = og * tanh_fast(c);
      __bf16 hb = (__bf16)h;
      h2_s[(r + 8 * half) * HDIM + n0 + nl] = hb;
      h2_seq[((long long)t * BSZ + (r + 8 * half)) * HDIM + n0 + nl] = hb;
    }
    __syncthreads();   // new h2_s visible for next step
  }
}

// ---------------- output projection GEMM ----------------
// out[b, t, v] = h2_seq[t-1, b, :] . embed_W[v, :] + fc_b[v]   (t >= 1)
// M=16 (batch) x N=256 vocab per block; 8 waves x 2 N-tiles; K=512.

__global__ __launch_bounds__(256) void proj_kernel(
    const __bf16* __restrict__ h2_seq,  // (255, 16, 512)
    const __bf16* __restrict__ embW,    // (32000, 512) bf16
    const float*  __restrict__ fc_b,    // (32000)
    float* __restrict__ out)            // (16, 256, 32000)
{
  __shared__ __align__(16) __bf16 A_s[BSZ * HDIM];   // 16 KB
  const int tTile = blockIdx.x;        // 0..254  -> t = tTile+1
  const int vChunk = blockIdx.y;       // 0..124

  {
    const int4* src = (const int4*)(h2_seq + (long long)tTile * BSZ * HDIM);
    int4* dst = (int4*)A_s;
    for (int i = threadIdx.x; i < (BSZ * HDIM) / 8; i += blockDim.x) dst[i] = src[i];
  }
  __syncthreads();

  const int lane = threadIdx.x & 31;
  const int wave = threadIdx.x >> 5;   // 0..7
  const int nl = lane & 15;
  const int half = lane >> 4;
  const int v0 = vChunk * 256 + wave * 32;

  v8f acc0 = {}; v8f acc1 = {};
  for (int k0 = 0; k0 < HDIM; k0 += 32) {
    v16bf a  = load_a(A_s, k0, lane);
    v16bf b0 = load_b(embW, v0,      k0, lane);
    v16bf b1 = load_b(embW, v0 + 16, k0, lane);
    acc0 = __builtin_amdgcn_wmma_f32_16x16x32_bf16(false, a, false, b0, (short)0, acc0, false, false);
    acc1 = __builtin_amdgcn_wmma_f32_16x16x32_bf16(false, a, false, b1, (short)0, acc1, false, false);
  }

  const float bias0 = fc_b[v0 + nl];
  const float bias1 = fc_b[v0 + 16 + nl];
  const int t = tTile + 1;
  #pragma unroll
  for (int r = 0; r < 8; ++r) {
    int b = r + 8 * half;
    long long base = ((long long)b * TLEN + t) * (long long)VOC;
    __builtin_nontemporal_store(acc0[r] + bias0, &out[base + v0 + nl]);
    __builtin_nontemporal_store(acc1[r] + bias1, &out[base + v0 + 16 + nl]);
  }
}

// ---------------- launcher ----------------

extern "C" void kernel_launch(void* const* d_in, const int* in_sizes, int n_in,
                              void* d_out, int out_size, void* d_ws, size_t ws_size,
                              hipStream_t stream) {
  const int*   words   = (const int*)d_in[0];
  const float* embed_W = (const float*)d_in[1];
  const float* fc_b    = (const float*)d_in[2];
  const float* w_ih1   = (const float*)d_in[3];
  const float* w_hh1   = (const float*)d_in[4];
  const float* b_ih1   = (const float*)d_in[5];
  const float* b_hh1   = (const float*)d_in[6];
  const float* w_ih2   = (const float*)d_in[7];
  const float* w_hh2   = (const float*)d_in[8];
  const float* b_ih2   = (const float*)d_in[9];
  const float* b_hh2   = (const float*)d_in[10];
  float* out = (float*)d_out;
  (void)in_sizes; (void)n_in; (void)out_size; (void)ws_size;

  char* ws = (char*)d_ws;
  size_t off = 0;
  auto alloc = [&](size_t bytes) -> char* {
    char* p = ws + off;
    off = (off + bytes + 255) & ~(size_t)255;
    return p;
  };
  __bf16* embW_bf  = (__bf16*)alloc((size_t)VOC * HDIM * 2);        // 32.8 MB
  __bf16* w_hh1_bf = (__bf16*)alloc((size_t)G4 * HDIM * 2);         //  2.1 MB
  __bf16* w_ih2_bf = (__bf16*)alloc((size_t)G4 * HDIM * 2);
  __bf16* w_hh2_bf = (__bf16*)alloc((size_t)G4 * HDIM * 2);
  __bf16* h2_seq   = (__bf16*)alloc((size_t)(TLEN - 1) * BSZ * HDIM * 2); // 4.2 MB
  float*  xg       = (float*)alloc((size_t)BSZ * G4 * 4);
  float*  bias2    = (float*)alloc((size_t)G4 * 4);

  const int nEmb = VOC * HDIM;
  const int nW   = G4 * HDIM;
  cvt_f32_bf16<<<(nEmb + 255) / 256, 256, 0, stream>>>(embed_W, embW_bf, nEmb);
  cvt_f32_bf16<<<(nW + 255) / 256, 256, 0, stream>>>(w_hh1, w_hh1_bf, nW);
  cvt_f32_bf16<<<(nW + 255) / 256, 256, 0, stream>>>(w_ih2, w_ih2_bf, nW);
  cvt_f32_bf16<<<(nW + 255) / 256, 256, 0, stream>>>(w_hh2, w_hh2_bf, nW);
  xgates_kernel<<<(BSZ * G4 + 255) / 256, 256, 0, stream>>>(words, embed_W, w_ih1, b_ih1, b_hh1, xg);
  bias2_kernel<<<(G4 + 255) / 256, 256, 0, stream>>>(b_ih2, b_hh2, bias2);
  zero_t0_kernel<<<(BSZ * VOC + 255) / 256, 256, 0, stream>>>(out);

  lstm_seq_kernel<<<1, 1024, 0, stream>>>(w_hh1_bf, w_ih2_bf, w_hh2_bf, xg, bias2, h2_seq);

  dim3 pg(TLEN - 1, VOC / 256);   // 255 x 125 blocks
  proj_kernel<<<pg, 256, 0, stream>>>(h2_seq, embW_bf, fc_b, out);
}
